// WassersteinDistanceLoss_48885317763827
// MI455X (gfx1250) — compile-verified, matrix operand order checked by reading, not run on hardware
//
#include <hip/hip_runtime.h>

typedef _Float16 half_t;
typedef _Float16 v16h __attribute__((ext_vector_type(16)));
typedef _Float16 v8h  __attribute__((ext_vector_type(8)));
typedef float    v8f  __attribute__((ext_vector_type(8)));

#define BS 64
#define NN 512
#define MMC 512
#define DD 768
#define TOT (BS * NN * MMC)                 // 16,777,216 elements
#define NEG2LOG2E (-2.885390081777927f)     // -2 * log2(e); A = exp(-2C) = exp2(C * this)

// ---------- helpers ----------
__device__ __forceinline__ unsigned umin2(unsigned a, unsigned b) { return a < b ? a : b; }
__device__ __forceinline__ unsigned umax2(unsigned a, unsigned b) { return a > b ? a : b; }

__device__ __forceinline__ float fastexp2(float x) { return exp2f(x); }  // -> v_exp_f32

// monotone float -> uint key (total order preserved for all finite floats)
__device__ __forceinline__ unsigned fkey(float f) {
  unsigned u = __float_as_uint(f);
  return (u & 0x80000000u) ? ~u : (u ^ 0x80000000u);
}
__device__ __forceinline__ float fdec(unsigned u) {
  return (u & 0x80000000u) ? __uint_as_float(u ^ 0x80000000u) : __uint_as_float(~u);
}

__device__ __forceinline__ float wave_sum(float s) {
  #pragma unroll
  for (int off = 16; off; off >>= 1) s += __shfl_xor(s, off, 32);
  return s;
}

// ---------- init: sigma=1/m, min/max seeds, zero output ----------
__global__ void k_init(float* __restrict__ sigma, unsigned* __restrict__ mm, float* __restrict__ out) {
  int i = blockIdx.x * blockDim.x + threadIdx.x;
  if (i < BS * MMC) sigma[i] = 1.0f / (float)MMC;
  if (i == 0) { mm[0] = 0xFFFFFFFFu; mm[1] = 0u; out[0] = 0.0f; }
}

// ---------- row-normalize fp32 -> f16, one wave32 per row of 768 ----------
__global__ void k_normalize(const float* __restrict__ in, half_t* __restrict__ out) {
  int wave = threadIdx.x >> 5;
  int lane = threadIdx.x & 31;
  int row = blockIdx.x * 8 + wave;
  const float* p = in + (size_t)row * DD;
  float vals[24];
  float s = 0.0f;
  #pragma unroll
  for (int j = 0; j < 24; ++j) { float v = p[lane + j * 32]; vals[j] = v; s += v * v; }
  s = wave_sum(s);
  float inv = 1.0f / (sqrtf(s) + 1e-12f);
  half_t* q = out + (size_t)row * DD;
  #pragma unroll
  for (int j = 0; j < 24; ++j) q[lane + j * 32] = (half_t)(vals[j] * inv);
}

// ---------- WMMA fragment load (16x32 f16 tile, ISA A/B layout) ----------
__device__ __forceinline__ v16h load_frag(const half_t* __restrict__ base, int row0, int k0, int lane) {
  int r = lane & 15;
  int h = lane >> 4;
  const half_t* p = base + (size_t)(row0 + r) * DD + k0 + h * 8;
  v8h lo = *(const v8h*)p;          // K = k0 + h*8 .. +7      -> VGPR 0..3
  v8h hi = *(const v8h*)(p + 16);   // K = k0 + 16 + h*8 .. +7 -> VGPR 4..7
  return __builtin_shufflevector(lo, hi, 0,1,2,3,4,5,6,7,8,9,10,11,12,13,14,15);
}

// ---------- batched C = 1 - Xn @ Yn^T via v_wmma_f32_16x16x32_f16 ----------
// block = 256 threads = 8 waves arranged 4(M) x 2(N); wave tile 32x32 (2x2 WMMA)
// grid = (512/64, 512/128, BS)
__global__ void k_gemm(const half_t* __restrict__ X, const half_t* __restrict__ Y,
                       float* __restrict__ C) {
  int lane = threadIdx.x & 31;
  int w = threadIdx.x >> 5;
  int wm = w >> 1;      // 0..3
  int wn = w & 1;       // 0..1
  int b = blockIdx.z;
  int r0 = blockIdx.y * 128 + wm * 32;
  int c0 = blockIdx.x * 64 + wn * 32;
  const half_t* Xb = X + (size_t)b * NN * DD;
  const half_t* Yb = Y + (size_t)b * MMC * DD;

  v8f acc00 = {}, acc01 = {}, acc10 = {}, acc11 = {};
  for (int k0 = 0; k0 < DD; k0 += 32) {
    v16h a0 = load_frag(Xb, r0,      k0, lane);
    v16h a1 = load_frag(Xb, r0 + 16, k0, lane);
    v16h b0 = load_frag(Yb, c0,      k0, lane);
    v16h b1 = load_frag(Yb, c0 + 16, k0, lane);
    acc00 = __builtin_amdgcn_wmma_f32_16x16x32_f16(false, a0, false, b0, (short)0, acc00, false, false);
    acc01 = __builtin_amdgcn_wmma_f32_16x16x32_f16(false, a0, false, b1, (short)0, acc01, false, false);
    acc10 = __builtin_amdgcn_wmma_f32_16x16x32_f16(false, a1, false, b0, (short)0, acc10, false, false);
    acc11 = __builtin_amdgcn_wmma_f32_16x16x32_f16(false, a1, false, b1, (short)0, acc11, false, false);
  }

  // C/D layout: VGPR v, lane L -> row = v + (L/16)*8, col = L%16
  float* Cb = C + (size_t)b * NN * MMC;
  int cr = lane >> 4;
  int cc = lane & 15;
  #pragma unroll
  for (int v = 0; v < 8; ++v) {
    int rA = r0 + v + cr * 8;
    int rB = r0 + 16 + v + cr * 8;
    Cb[(size_t)rA * MMC + c0 + cc]      = 1.0f - acc00[v];
    Cb[(size_t)rA * MMC + c0 + 16 + cc] = 1.0f - acc01[v];
    Cb[(size_t)rB * MMC + c0 + cc]      = 1.0f - acc10[v];
    Cb[(size_t)rB * MMC + c0 + 16 + cc] = 1.0f - acc11[v];
  }
}

// ---------- global min/max of C ----------
__global__ void k_minmax(const float* __restrict__ C, unsigned* __restrict__ mm) {
  unsigned kmin = 0xFFFFFFFFu, kmax = 0u;
  for (size_t i = (size_t)blockIdx.x * blockDim.x + threadIdx.x; i < (size_t)TOT;
       i += (size_t)gridDim.x * blockDim.x) {
    unsigned k = fkey(C[i]);
    kmin = umin2(kmin, k);
    kmax = umax2(kmax, k);
  }
  #pragma unroll
  for (int off = 16; off; off >>= 1) {
    kmin = umin2(kmin, (unsigned)__shfl_xor((int)kmin, off, 32));
    kmax = umax2(kmax, (unsigned)__shfl_xor((int)kmax, off, 32));
  }
  __shared__ unsigned smin[8], smax[8];
  int w = threadIdx.x >> 5, lane = threadIdx.x & 31;
  if (lane == 0) { smin[w] = kmin; smax[w] = kmax; }
  __syncthreads();
  if (threadIdx.x == 0) {
    unsigned a = smin[0], bm = smax[0];
    #pragma unroll
    for (int i = 1; i < 8; ++i) { a = umin2(a, smin[i]); bm = umax2(bm, smax[i]); }
    atomicMin(&mm[0], a);
    atomicMax(&mm[1], bm);
  }
}

// ---------- threshold + relu; init T = 1 ----------
__global__ void k_thresh(float* __restrict__ C, float* __restrict__ T,
                         const unsigned* __restrict__ mm) {
  float mn = fdec(mm[0]);
  float mx = fdec(mm[1]);
  float th = mn + 0.1f * (mx - mn);
  size_t i = (size_t)blockIdx.x * blockDim.x + threadIdx.x;
  if (i < (size_t)TOT) {
    float c = fmaxf(C[i] - th, 0.0f);
    C[i] = c;
    T[i] = 1.0f;
  }
}

// ---------- fused IPOT row pass ----------
// If apply: T <- T * A * delta_prev_i * sigma_j (lazy update from previous iter)
// Then:     delta_i = 1/(n * sum_j A*T*sigma_j)
// One wave32 per row; float4 vector loads/stores.
__global__ void k_rowf(const float* __restrict__ C, float* __restrict__ T,
                       const float* __restrict__ sigma, float* __restrict__ delta,
                       int apply) {
  int w = threadIdx.x >> 5, lane = threadIdx.x & 31;
  int rid = blockIdx.x * 8 + w;          // b*NN + i
  int b = rid >> 9;
  const float4* Cr = (const float4*)(C + (size_t)rid * MMC);
  float4*       Tr = (float4*)(T + (size_t)rid * MMC);
  const float4* sg = (const float4*)(sigma + b * MMC);
  float dp = apply ? delta[rid] : 1.0f;
  float s = 0.0f;
  #pragma unroll
  for (int j = 0; j < 4; ++j) {
    int v = lane + j * 32;               // float4 index 0..127
    float4 c4 = Cr[v];
    float4 t4 = Tr[v];
    float4 s4 = sg[v];
    float a0 = fastexp2(c4.x * NEG2LOG2E);
    float a1 = fastexp2(c4.y * NEG2LOG2E);
    float a2 = fastexp2(c4.z * NEG2LOG2E);
    float a3 = fastexp2(c4.w * NEG2LOG2E);
    if (apply) {
      t4.x *= a0 * dp * s4.x;
      t4.y *= a1 * dp * s4.y;
      t4.z *= a2 * dp * s4.z;
      t4.w *= a3 * dp * s4.w;
      Tr[v] = t4;
    }
    s += a0 * t4.x * s4.x + a1 * t4.y * s4.y + a2 * t4.z * s4.z + a3 * t4.w * s4.w;
  }
  s = wave_sum(s);
  if (lane == 0) delta[rid] = 1.0f / ((float)NN * s);
}

// ---------- IPOT col pass, split-rows partials ----------
// grid = (2, 4, BS): x = column half (256 cols), y = row chunk (128 rows), z = batch
__global__ void k_colp(const float* __restrict__ C, const float* __restrict__ T,
                       const float* __restrict__ delta, float* __restrict__ part) {
  __shared__ float sd[128];
  int b = blockIdx.z;
  int chunk = blockIdx.y;
  int j = (blockIdx.x << 8) + threadIdx.x;
  int i0 = chunk * 128;
  if (threadIdx.x < 128) sd[threadIdx.x] = delta[b * NN + i0 + threadIdx.x];
  __syncthreads();
  const float* Cb = C + (size_t)b * NN * MMC;
  const float* Tb = T + (size_t)b * NN * MMC;
  float s = 0.0f;
  for (int i = 0; i < 128; ++i) {
    size_t idx = (size_t)(i0 + i) * MMC + j;
    float A = fastexp2(Cb[idx] * NEG2LOG2E);
    s += A * Tb[idx] * sd[i];
  }
  part[((size_t)(b * 4 + chunk)) * MMC + j] = s;
}

// ---------- combine partials: sigma_j = 1/(m * sum_chunks part) ----------
__global__ void k_sigma(const float* __restrict__ part, float* __restrict__ sigma) {
  int i = blockIdx.x * blockDim.x + threadIdx.x;   // b*MMC + j
  if (i >= BS * MMC) return;
  int b = i >> 9;
  int j = i & (MMC - 1);
  float s = 0.0f;
  #pragma unroll
  for (int c = 0; c < 4; ++c) s += part[((size_t)(b * 4 + c)) * MMC + j];
  sigma[i] = 1.0f / ((float)MMC * s);
}

// ---------- final: apply pending update, out = -(1/BS) * sum C*T_final ----------
__global__ void k_final(const float* __restrict__ C, const float* __restrict__ T,
                        const float* __restrict__ delta, const float* __restrict__ sigma,
                        float* __restrict__ out) {
  float s = 0.0f;
  for (size_t idx = (size_t)blockIdx.x * blockDim.x + threadIdx.x; idx < (size_t)TOT;
       idx += (size_t)gridDim.x * blockDim.x) {
    int j = (int)(idx & (MMC - 1));
    int ri = (int)(idx >> 9);      // b*NN + i
    int b = ri >> 9;
    float c = C[idx];
    float A = fastexp2(c * NEG2LOG2E);
    float tfin = T[idx] * A * delta[ri] * sigma[b * MMC + j];
    s += c * tfin;
  }
  s = wave_sum(s);
  __shared__ float sw[8];
  int w = threadIdx.x >> 5, lane = threadIdx.x & 31;
  if (lane == 0) sw[w] = s;
  __syncthreads();
  if (threadIdx.x == 0) {
    float t = 0.0f;
    #pragma unroll
    for (int i = 0; i < 8; ++i) t += sw[i];
    atomicAdd(out, -t * (1.0f / (float)BS));
  }
}

// ---------- workspace layout ----------
//  [0, 50.3MB)   Xh (f16)        \  after k_gemm these are dead;
//  [50.3, 100.7) Yh (f16)        /  T (67.1MB fp32) aliases offset 0
//  [100.7MB)     C  (fp32, 67.1MB)
//  then sigma (128KB), delta (128KB), minmax (8B, padded), part (512KB)
#define XH_BYTES ((size_t)BS * NN * DD * 2)       // 50,331,648
#define C_OFF    (2 * XH_BYTES)                   // 100,663,296
#define C_BYTES  ((size_t)TOT * 4)                // 67,108,864
#define SIG_OFF  (C_OFF + C_BYTES)
#define DEL_OFF  (SIG_OFF + (size_t)BS * MMC * 4)
#define MM_OFF   (DEL_OFF + (size_t)BS * NN * 4)
#define PART_OFF (MM_OFF + 256)

extern "C" void kernel_launch(void* const* d_in, const int* in_sizes, int n_in,
                              void* d_out, int out_size, void* d_ws, size_t ws_size,
                              hipStream_t stream) {
  const float* X = (const float*)d_in[0];
  const float* Y = (const float*)d_in[1];
  float* out = (float*)d_out;
  char* ws = (char*)d_ws;

  half_t* Xh = (half_t*)ws;
  half_t* Yh = (half_t*)(ws + XH_BYTES);
  float* T     = (float*)ws;               // aliases Xh/Yh (dead after GEMM)
  float* C     = (float*)(ws + C_OFF);
  float* sigma = (float*)(ws + SIG_OFF);
  float* delta = (float*)(ws + DEL_OFF);
  unsigned* mm = (unsigned*)(ws + MM_OFF);
  float* part  = (float*)(ws + PART_OFF);

  k_init<<<(BS * MMC + 255) / 256, 256, 0, stream>>>(sigma, mm, out);
  k_normalize<<<BS * NN / 8, 256, 0, stream>>>(X, Xh);
  k_normalize<<<BS * MMC / 8, 256, 0, stream>>>(Y, Yh);
  k_gemm<<<dim3(MMC / 64, NN / 128, BS), 256, 0, stream>>>(Xh, Yh, C);
  k_minmax<<<4096, 256, 0, stream>>>(C, mm);
  k_thresh<<<TOT / 256, 256, 0, stream>>>(C, T, mm);
  for (int it = 0; it < 30; ++it) {
    k_rowf<<<BS * NN / 8, 256, 0, stream>>>(C, T, sigma, delta, it > 0 ? 1 : 0);
    k_colp<<<dim3(2, 4, BS), 256, 0, stream>>>(C, T, delta, part);
    k_sigma<<<BS * MMC / 256, 256, 0, stream>>>(part, sigma);
  }
  k_final<<<4096, 256, 0, stream>>>(C, T, delta, sigma, out);
}
